// MechanicalPropertiesPredictor_49795850830265
// MI455X (gfx1250) — matrix operand order repaired and checked by reading, not compiled, not run on hardware
//
#include <hip/hip_runtime.h>
#include <hip/hip_bf16.h>
#include <math.h>

typedef __bf16 bf16_t;
typedef __attribute__((ext_vector_type(16))) __bf16 v16bf;
typedef __attribute__((ext_vector_type(8)))  __bf16 v8bf;
typedef __attribute__((ext_vector_type(8)))  float  v8f;

#define NNODES 50000
#define NEDGES 800000
#define NGRAPH 256
#define HID    128
#define SC_STRIDE 392   // 384 + 8 pad (bank-conflict-free b128 A-frag loads)
#define HB_STRIDE 136   // 128 + 8 pad
#define CT_STRIDE 264   // 256 + 8 pad

// ---------------------------------------------------------------------------
// WMMA helpers (CDNA5 wave32, D = A(16x32 bf16) * B(32x16 bf16) + C(f32))
// ---------------------------------------------------------------------------
__device__ __forceinline__ v8f wmma_bf16(v16bf a, v16bf b, v8f c) {
  return __builtin_amdgcn_wmma_f32_16x16x32_bf16(false, a, false, b, (short)0, c,
                                                 false, false);
}

union BfFrag { v16bf v; v8bf h[2]; };

// A-frag (16x32, M=edge rows in LDS row-major): lane<16 -> M=lane, K chunks
// {0..7,16..23}; lane>=16 -> M=lane-16, K chunks {8..15,24..31}. Two b128 loads.
// B-frag pre-packed in the same k order: lane l element i <-> K = (l>=16)*8 + (i&7) + (i>>3)*16, N = l&15.
template<int STRIDE, int KSTEPS>
__device__ __forceinline__ v8f gemm_tile(const bf16_t* smem, const bf16_t* Wp, v8f acc) {
  const int lane = threadIdx.x & 31;
  const int wv   = threadIdx.x >> 5;                 // N-tile (0..7)
  const int m    = lane & 15;
  const int koff = (lane >> 4) << 3;
  const bf16_t* arow = smem + m * STRIDE + koff;
  const bf16_t* bptr = Wp + ((size_t)(wv * KSTEPS) * 32 + lane) * 16;
#pragma unroll
  for (int s = 0; s < KSTEPS; ++s) {
    BfFrag a, b;
    a.h[0] = *(const v8bf*)(arow + s * 32);
    a.h[1] = *(const v8bf*)(arow + s * 32 + 16);
    b.h[0] = *(const v8bf*)(bptr);
    b.h[1] = *(const v8bf*)(bptr + 8);
    bptr += 512;
    acc = wmma_bf16(a.v, b.v, acc);
  }
  return acc;
}

// ---------------------------------------------------------------------------
// Weight repack: W[K][N] f32 row-major -> bf16 B-fragment stream
// p = ((ntile*ksteps + s)*32 + lane)*16 + i ; K = s*32 + (lane>=16)*8 + (i&7) + (i>>3)*16
// ---------------------------------------------------------------------------
__global__ __launch_bounds__(256) void pack_wmma_b(const float* __restrict__ W,
                                                   bf16_t* __restrict__ dst,
                                                   int K, int N) {
  int p = blockIdx.x * 256 + threadIdx.x;
  int ksteps = K >> 5;
  int total  = (K >> 5) * (N >> 4) * 512;
  if (p >= total) return;
  int i  = p & 15;
  int l  = (p >> 4) & 31;
  int g  = p >> 9;
  int s  = g % ksteps;
  int tl = g / ksteps;
  int k  = s * 32 + ((l >> 4) << 3) + (i & 7) + ((i >> 3) << 4);
  int n  = tl * 16 + (l & 15);
  dst[p] = (bf16_t)W[(size_t)k * N + n];
}

// ---------------------------------------------------------------------------
// Node encoder: h = relu(x @ W(38x128) + b), also bf16 shadow copy
// ---------------------------------------------------------------------------
__global__ __launch_bounds__(128) void node_encode(const float* __restrict__ x,
                                                   const float* __restrict__ W,
                                                   const float* __restrict__ b,
                                                   float* __restrict__ h,
                                                   bf16_t* __restrict__ hb) {
  __shared__ float sx[40];
  const int n = blockIdx.x, c = threadIdx.x;
  if (c < 38) sx[c] = x[(size_t)n * 38 + c];
  __syncthreads();
  float acc = b[c];
#pragma unroll 2
  for (int k = 0; k < 38; ++k) acc += sx[k] * W[k * 128 + c];
  acc = fmaxf(acc, 0.0f);
  h[(size_t)n * 128 + c]  = acc;
  hb[(size_t)n * 128 + c] = (bf16_t)acc;
}

// ---------------------------------------------------------------------------
// Shared staging for edge kernels: cat = [h[dst] | h[src] | e(recomputed)] bf16
// ---------------------------------------------------------------------------
__device__ __forceinline__ void stage_cat(bf16_t* sc, float* sWe, float* sEA,
                                          const bf16_t* hb, const float* edge_attr,
                                          const int* srcI, const int* dstI,
                                          const float* We, const float* be, int e0) {
  const int t = threadIdx.x;
  for (int i = t; i < 7 * 128; i += 256) sWe[i] = We[i];
  if (t < 112) {
    int j = t / 7, k = t - j * 7;
    sEA[j * 8 + k] = edge_attr[(size_t)(e0 + j) * 7 + k];
  }
  __syncthreads();
  const int j = t >> 4, seg = t & 15;
  {  // gather h rows (16 halves = 2x uint4 per thread)
    int node = (seg < 8) ? dstI[e0 + j] : srcI[e0 + j];
    int c0 = (seg & 7) * 16;
    const uint4* sp = (const uint4*)(hb + (size_t)node * HID + c0);
    uint4* dp = (uint4*)(sc + j * SC_STRIDE + ((seg < 8) ? 0 : HID) + c0);
    dp[0] = sp[0];
    dp[1] = sp[1];
  }
  {  // recompute e channels (7 MACs each), 8 channels per thread
    int cc0 = (t & 15) * 8;
#pragma unroll
    for (int c = cc0; c < cc0 + 8; ++c) {
      float acc = be[c];
#pragma unroll
      for (int k = 0; k < 7; ++k) acc += sEA[j * 8 + k] * sWe[k * 128 + c];
      sc[j * SC_STRIDE + 256 + c] = (bf16_t)fmaxf(acc, 0.0f);
    }
  }
  __syncthreads();
}

// ---------------------------------------------------------------------------
// Edge pass A: attn logits = leakyrelu(cat@W1+b1, 0.2) @ W2 + b2  -> [E,4]
// ---------------------------------------------------------------------------
__global__ __launch_bounds__(256) void edge_pass_a(
    const bf16_t* __restrict__ hb, const float* __restrict__ edge_attr,
    const int* __restrict__ srcI, const int* __restrict__ dstI,
    const float* __restrict__ We, const float* __restrict__ be,
    const bf16_t* __restrict__ W1p, const float* __restrict__ b1,
    const float* __restrict__ W2, const float* __restrict__ b2,
    float* __restrict__ logits) {
  __shared__ __align__(16) bf16_t sc[16 * SC_STRIDE];
  __shared__ float sWe[7 * 128];
  __shared__ float sEA[16 * 8];
  __shared__ float sW2[512];
  __shared__ __align__(16) float sHid[16][136];
  const int t = threadIdx.x;
  const int e0 = blockIdx.x * 16;
  for (int i = t; i < 512; i += 256) sW2[i] = W2[i];
  stage_cat(sc, sWe, sEA, hb, edge_attr, srcI, dstI, We, be, e0);

  v8f acc = {0.f, 0.f, 0.f, 0.f, 0.f, 0.f, 0.f, 0.f};
  acc = gemm_tile<SC_STRIDE, 12>(sc, W1p, acc);

  const int lane = t & 31, wv = t >> 5;
  const int n  = wv * 16 + (lane & 15);
  const int mb = (lane >> 4) * 8;
  const float bias = b1[n];
#pragma unroll
  for (int r = 0; r < 8; ++r) {
    float v = acc[r] + bias;
    sHid[mb + r][n] = (v > 0.f) ? v : 0.2f * v;   // leaky_relu 0.2
  }
  __syncthreads();
  if (t < 64) {
    const int j = t >> 2, hd = t & 3;
    float a2 = b2[hd];
#pragma unroll 8
    for (int c = 0; c < 128; c += 4) {
      float4 hv = *(const float4*)&sHid[j][c];
      a2 += hv.x * sW2[(c + 0) * 4 + hd] + hv.y * sW2[(c + 1) * 4 + hd] +
            hv.z * sW2[(c + 2) * 4 + hd] + hv.w * sW2[(c + 3) * 4 + hd];
    }
    logits[(size_t)(e0 + j) * 4 + hd] = a2;
  }
}

// ---------------------------------------------------------------------------
// Global softmax reductions over E edges, per head (4 heads)
// ---------------------------------------------------------------------------
__global__ __launch_bounds__(256) void attn_partial_max(const float* __restrict__ lg,
                                                        float* __restrict__ pmax, int E) {
  __shared__ float s[1024];
  const int t = threadIdx.x;
  float m0 = -1e30f, m1 = -1e30f, m2 = -1e30f, m3 = -1e30f;
  for (int e = blockIdx.x * 256 + t; e < E; e += gridDim.x * 256) {
    float4 l = *(const float4*)&lg[(size_t)e * 4];
    m0 = fmaxf(m0, l.x); m1 = fmaxf(m1, l.y); m2 = fmaxf(m2, l.z); m3 = fmaxf(m3, l.w);
  }
  s[t] = m0; s[256 + t] = m1; s[512 + t] = m2; s[768 + t] = m3;
  __syncthreads();
  for (int off = 128; off; off >>= 1) {
    if (t < off) {
      s[t]       = fmaxf(s[t], s[t + off]);
      s[256 + t] = fmaxf(s[256 + t], s[256 + t + off]);
      s[512 + t] = fmaxf(s[512 + t], s[512 + t + off]);
      s[768 + t] = fmaxf(s[768 + t], s[768 + t + off]);
    }
    __syncthreads();
  }
  if (t < 4) pmax[blockIdx.x * 4 + t] = s[t * 256];
}

__global__ __launch_bounds__(256) void attn_final_max(const float* __restrict__ part,
                                                      float* __restrict__ gmax) {
  __shared__ float s[1024];
  const int t = threadIdx.x;
  float4 v = *(const float4*)&part[(size_t)t * 4];
  s[t] = v.x; s[256 + t] = v.y; s[512 + t] = v.z; s[768 + t] = v.w;
  __syncthreads();
  for (int off = 128; off; off >>= 1) {
    if (t < off) {
      s[t]       = fmaxf(s[t], s[t + off]);
      s[256 + t] = fmaxf(s[256 + t], s[256 + t + off]);
      s[512 + t] = fmaxf(s[512 + t], s[512 + t + off]);
      s[768 + t] = fmaxf(s[768 + t], s[768 + t + off]);
    }
    __syncthreads();
  }
  if (t < 4) gmax[t] = s[t * 256];
}

__global__ __launch_bounds__(256) void attn_partial_sum(const float* __restrict__ lg,
                                                        const float* __restrict__ gmax,
                                                        float* __restrict__ psum, int E) {
  __shared__ float s[1024];
  const int t = threadIdx.x;
  const float g0 = gmax[0], g1 = gmax[1], g2 = gmax[2], g3 = gmax[3];
  float s0 = 0, s1 = 0, s2 = 0, s3 = 0;
  for (int e = blockIdx.x * 256 + t; e < E; e += gridDim.x * 256) {
    float4 l = *(const float4*)&lg[(size_t)e * 4];
    s0 += expf(l.x - g0); s1 += expf(l.y - g1);
    s2 += expf(l.z - g2); s3 += expf(l.w - g3);
  }
  s[t] = s0; s[256 + t] = s1; s[512 + t] = s2; s[768 + t] = s3;
  __syncthreads();
  for (int off = 128; off; off >>= 1) {
    if (t < off) {
      s[t] += s[t + off]; s[256 + t] += s[256 + t + off];
      s[512 + t] += s[512 + t + off]; s[768 + t] += s[768 + t + off];
    }
    __syncthreads();
  }
  if (t < 4) psum[blockIdx.x * 4 + t] = s[t * 256];
}

__global__ __launch_bounds__(256) void attn_final_sum(const float* __restrict__ part,
                                                      float* __restrict__ gsuminv) {
  __shared__ float s[1024];
  const int t = threadIdx.x;
  float4 v = *(const float4*)&part[(size_t)t * 4];
  s[t] = v.x; s[256 + t] = v.y; s[512 + t] = v.z; s[768 + t] = v.w;
  __syncthreads();
  for (int off = 128; off; off >>= 1) {
    if (t < off) {
      s[t] += s[t + off]; s[256 + t] += s[256 + t + off];
      s[512 + t] += s[512 + t + off]; s[768 + t] += s[768 + t + off];
    }
    __syncthreads();
  }
  if (t < 4) gsuminv[t] = 1.0f / s[t * 256];
}

// ---------------------------------------------------------------------------
// Edge pass B: m = relu(cat@W1+b1)@W2+b2 ; agg[dst] += a[e]*m  (f32 atomics)
// ---------------------------------------------------------------------------
__global__ __launch_bounds__(256) void edge_pass_b(
    const bf16_t* __restrict__ hb, const float* __restrict__ edge_attr,
    const int* __restrict__ srcI, const int* __restrict__ dstI,
    const float* __restrict__ We, const float* __restrict__ be,
    const bf16_t* __restrict__ W1p, const float* __restrict__ b1,
    const bf16_t* __restrict__ W2p, const float* __restrict__ b2,
    const float* __restrict__ logits, const float* __restrict__ gmax,
    const float* __restrict__ gsuminv, float* __restrict__ agg) {
  __shared__ __align__(16) bf16_t sc[16 * SC_STRIDE];
  __shared__ float sWe[7 * 128];
  __shared__ float sEA[16 * 8];
  __shared__ __align__(16) bf16_t sHidB[16 * HB_STRIDE];
  __shared__ float sA[16];
  __shared__ int sDst[16];
  const int t = threadIdx.x;
  const int e0 = blockIdx.x * 16;
  if (t < 16) {
    sDst[t] = dstI[e0 + t];
    float4 lg = *(const float4*)&logits[(size_t)(e0 + t) * 4];
    float a = expf(lg.x - gmax[0]) * gsuminv[0] + expf(lg.y - gmax[1]) * gsuminv[1] +
              expf(lg.z - gmax[2]) * gsuminv[2] + expf(lg.w - gmax[3]) * gsuminv[3];
    sA[t] = 0.25f * a;   // mean over heads
  }
  stage_cat(sc, sWe, sEA, hb, edge_attr, srcI, dstI, We, be, e0);

  v8f acc = {0.f, 0.f, 0.f, 0.f, 0.f, 0.f, 0.f, 0.f};
  acc = gemm_tile<SC_STRIDE, 12>(sc, W1p, acc);

  const int lane = t & 31, wv = t >> 5;
  const int n  = wv * 16 + (lane & 15);
  const int mb = (lane >> 4) * 8;
  const float bias = b1[n];
#pragma unroll
  for (int r = 0; r < 8; ++r)
    sHidB[(mb + r) * HB_STRIDE + n] = (bf16_t)fmaxf(acc[r] + bias, 0.0f);
  __syncthreads();

  v8f acc2 = {0.f, 0.f, 0.f, 0.f, 0.f, 0.f, 0.f, 0.f};
  acc2 = gemm_tile<HB_STRIDE, 4>(sHidB, W2p, acc2);
  const float b2v = b2[n];
#pragma unroll
  for (int r = 0; r < 8; ++r) {
    int m = mb + r;
    float v = (acc2[r] + b2v) * sA[m];
    atomicAdd(&agg[(size_t)sDst[m] * 128 + n], v);
  }
}

// ---------------------------------------------------------------------------
// Node update: out = relu([h|agg]@W1+b1)@W2+b2  (in-place into agg buffer)
// ---------------------------------------------------------------------------
__global__ __launch_bounds__(256) void node_update(
    const bf16_t* __restrict__ hb,
    const bf16_t* __restrict__ W1p, const float* __restrict__ b1,
    const bf16_t* __restrict__ W2p, const float* __restrict__ b2,
    float* __restrict__ agg) {
  __shared__ __align__(16) bf16_t scat[16 * CT_STRIDE];
  __shared__ __align__(16) bf16_t sHidB[16 * HB_STRIDE];
  const int t = threadIdx.x;
  const int n0 = blockIdx.x * 16;
  const int j = t >> 4, seg = t & 15;
  const size_t node = (size_t)(n0 + j);
  if (seg < 8) {
    int c0 = seg * 16;
    const uint4* sp = (const uint4*)(hb + node * 128 + c0);
    uint4* dp = (uint4*)(scat + j * CT_STRIDE + c0);
    dp[0] = sp[0];
    dp[1] = sp[1];
  } else {
    int c0 = (seg - 8) * 16;
    const float4* ap = (const float4*)(agg + node * 128 + c0);
#pragma unroll
    for (int q = 0; q < 4; ++q) {
      float4 v = ap[q];
      bf16_t* d = scat + j * CT_STRIDE + 128 + c0 + q * 4;
      d[0] = (bf16_t)v.x; d[1] = (bf16_t)v.y; d[2] = (bf16_t)v.z; d[3] = (bf16_t)v.w;
    }
  }
  __syncthreads();

  v8f acc = {0.f, 0.f, 0.f, 0.f, 0.f, 0.f, 0.f, 0.f};
  acc = gemm_tile<CT_STRIDE, 8>(scat, W1p, acc);
  const int lane = t & 31, wv = t >> 5;
  const int n  = wv * 16 + (lane & 15);
  const int mb = (lane >> 4) * 8;
  const float bias = b1[n];
#pragma unroll
  for (int r = 0; r < 8; ++r)
    sHidB[(mb + r) * HB_STRIDE + n] = (bf16_t)fmaxf(acc[r] + bias, 0.0f);
  __syncthreads();

  v8f acc2 = {0.f, 0.f, 0.f, 0.f, 0.f, 0.f, 0.f, 0.f};
  acc2 = gemm_tile<HB_STRIDE, 4>(sHidB, W2p, acc2);
  const float b2v = b2[n];
#pragma unroll
  for (int r = 0; r < 8; ++r)
    agg[(size_t)(n0 + mb + r) * 128 + n] = acc2[r] + b2v;
}

// ---------------------------------------------------------------------------
// GraphNorm column reductions (mean over all nodes, then var of o)
// ---------------------------------------------------------------------------
__global__ __launch_bounds__(128) void col_partial_sum(const float* __restrict__ out,
                                                       float* __restrict__ part, int N) {
  const int c = threadIdx.x, b = blockIdx.x;
  float s = 0.f;
  for (int n = b; n < N; n += gridDim.x) s += out[(size_t)n * 128 + c];
  part[b * 128 + c] = s;
}
__global__ __launch_bounds__(128) void col_final_mean(const float* __restrict__ part,
                                                      int nb, float* __restrict__ mean,
                                                      float invN) {
  const int c = threadIdx.x;
  float s = 0.f;
  for (int b = 0; b < nb; ++b) s += part[b * 128 + c];
  mean[c] = s * invN;
}
__global__ __launch_bounds__(128) void col_partial_var(const float* __restrict__ out,
                                                       const float* __restrict__ mean,
                                                       const float* __restrict__ ms,
                                                       float* __restrict__ part, int N) {
  const int c = threadIdx.x, b = blockIdx.x;
  const float mu = mean[c] * ms[c];
  float s = 0.f;
  for (int n = b; n < N; n += gridDim.x) {
    float o = out[(size_t)n * 128 + c] - mu;
    s += o * o;
  }
  part[b * 128 + c] = s;
}
__global__ __launch_bounds__(128) void col_final_var(const float* __restrict__ part,
                                                     int nb, float* __restrict__ var,
                                                     float invN) {
  const int c = threadIdx.x;
  float s = 0.f;
  for (int b = 0; b < nb; ++b) s += part[b * 128 + c];
  var[c] = s * invN;
}

// ---------------------------------------------------------------------------
// GraphNorm apply + residual + LayerNorm -> new h (f32 + bf16 shadow)
// ---------------------------------------------------------------------------
__global__ __launch_bounds__(256) void norm_residual_ln(
    const float* __restrict__ outb, const float* __restrict__ mean,
    const float* __restrict__ var, const float* __restrict__ gw,
    const float* __restrict__ gb, const float* __restrict__ ms,
    const float* __restrict__ lng, const float* __restrict__ lnb,
    float* __restrict__ h, bf16_t* __restrict__ hb) {
  __shared__ float red[256];
  const int t = threadIdx.x;
  const int j = t >> 7;
  const int c = t & 127;
  const size_t node = (size_t)blockIdx.x * 2 + j;
  const float mu = mean[c] * ms[c];
  float o = outb[node * 128 + c] - mu;
  float v = gw[c] * o * rsqrtf(var[c] + 1e-5f) + gb[c];
  float x = v + h[node * 128 + c];
  red[t] = x;
  __syncthreads();
  for (int off = 64; off; off >>= 1) {
    if (c < off) red[t] += red[t + off];
    __syncthreads();
  }
  const float m = red[j * 128] * (1.0f / 128.0f);
  __syncthreads();
  float d = x - m;
  red[t] = d * d;
  __syncthreads();
  for (int off = 64; off; off >>= 1) {
    if (c < off) red[t] += red[t + off];
    __syncthreads();
  }
  const float rv = rsqrtf(red[j * 128] * (1.0f / 128.0f) + 1e-5f);
  float hn = lng[c] * d * rv + lnb[c];
  h[node * 128 + c]  = hn;
  hb[node * 128 + c] = (bf16_t)hn;
}

// ---------------------------------------------------------------------------
// Pooling: per-graph sum / count / max (float max via order-preserving flip)
// ---------------------------------------------------------------------------
__global__ __launch_bounds__(256) void pool_nodes(const float* __restrict__ h,
                                                  const int* __restrict__ batch,
                                                  float* __restrict__ padd,
                                                  unsigned* __restrict__ pmaxu,
                                                  float* __restrict__ pcnt) {
  const int t = threadIdx.x;
  const size_t node = (size_t)blockIdx.x * 2 + (t >> 7);
  const int c = t & 127;
  const int g = batch[node];
  const float v = h[node * 128 + c];
  atomicAdd(&padd[g * 128 + c], v);
  unsigned bits = __float_as_uint(v);
  unsigned key = bits ^ (unsigned)(((int)bits >> 31) | 0x80000000);
  atomicMax(&pmaxu[g * 128 + c], key);
  if (c == 0) atomicAdd(&pcnt[g], 1.0f);
}

__global__ __launch_bounds__(256) void fill_u32(unsigned* __restrict__ p, unsigned v,
                                                int n) {
  int i = blockIdx.x * 256 + threadIdx.x;
  if (i < n) p[i] = v;
}

// ---------------------------------------------------------------------------
// Per-graph heads: hg=[mean|max|add] -> 3x MLP(384->128->64->1) -> sigmoid
// ---------------------------------------------------------------------------
struct HeadW { const float *w1, *b1, *w2, *b2, *w3, *b3; };

__global__ __launch_bounds__(128) void heads_kernel(const float* __restrict__ padd,
                                                    const unsigned* __restrict__ pmaxu,
                                                    const float* __restrict__ pcnt,
                                                    HeadW h0, HeadW h1, HeadW h2,
                                                    float* __restrict__ out) {
  __shared__ float hg[384];
  __shared__ float z1[128];
  __shared__ float z2[64];
  __shared__ float res[3];
  const int g = blockIdx.x, t = threadIdx.x;
  const float cnt = fmaxf(pcnt[g], 1.0f);
  const float add = padd[g * 128 + t];
  unsigned u = pmaxu[g * 128 + t];
  unsigned bits = (u & 0x80000000u) ? (u ^ 0x80000000u) : ~u;
  hg[t]        = add / cnt;
  hg[128 + t]  = __uint_as_float(bits);
  hg[256 + t]  = add;
  __syncthreads();
  const HeadW hw[3] = {h0, h1, h2};
#pragma unroll
  for (int hd = 0; hd < 3; ++hd) {
    float a = hw[hd].b1[t];
    for (int k = 0; k < 384; ++k) a += hg[k] * hw[hd].w1[k * 128 + t];
    z1[t] = fmaxf(a, 0.0f);
    __syncthreads();
    if (t < 64) {
      float a2 = hw[hd].b2[t];
      for (int k = 0; k < 128; ++k) a2 += z1[k] * hw[hd].w2[k * 64 + t];
      z2[t] = fmaxf(a2, 0.0f);
    }
    __syncthreads();
    if (t == 0) {
      float a3 = hw[hd].b3[0];
      for (int k = 0; k < 64; ++k) a3 += z2[k] * hw[hd].w3[k];
      res[hd] = 1.0f / (1.0f + expf(-a3));
    }
    __syncthreads();
  }
  if (t == 0) {
    out[g]       = res[0];
    out[256 + g] = res[1];
    out[512 + g] = res[2];
    out[768 + g] = 0.45f * res[0] + 0.35f * res[1] + 0.2f * res[2];
  }
}

// ---------------------------------------------------------------------------
// Host launcher
// ---------------------------------------------------------------------------
extern "C" void kernel_launch(void* const* d_in, const int* in_sizes, int n_in,
                              void* d_out, int out_size, void* d_ws, size_t ws_size,
                              hipStream_t stream) {
  (void)in_sizes; (void)n_in; (void)out_size; (void)ws_size;
  const int N = NNODES, E = NEDGES, G = NGRAPH;

  // Input mapping: depth-first insertion order of setup_inputs() dict (94 leaves).
  const float* x         = (const float*)d_in[0];
  const float* edge_attr = (const float*)d_in[1];
  const float* ne_w = (const float*)d_in[2];
  const float* ne_b = (const float*)d_in[3];
  const float* ee_w = (const float*)d_in[4];
  const float* ee_b = (const float*)d_in[5];
  auto L = [&](int l, int k) -> const float* { return (const float*)d_in[6 + l * 17 + k]; };
  // per-layer: 0 attn_l1_w 1 attn_l1_b 2 attn_l2_w 3 attn_l2_b 4 msg_l1_w 5 msg_l1_b
  //            6 msg_l2_w 7 msg_l2_b 8 upd_l1_w 9 upd_l1_b 10 upd_l2_w 11 upd_l2_b
  //            12 gn_w 13 gn_b 14 gn_ms 15 ln_g 16 ln_b
  const int* eidx  = (const int*)d_in[92];
  const int* batch = (const int*)d_in[93];
  const int* srcI = eidx;
  const int* dstI = eidx + E;

  // Workspace layout
  char* wsb = (char*)d_ws;
  size_t off = 0;
  auto alloc = [&](size_t bytes) -> void* {
    off = (off + 255) & ~(size_t)255;
    void* p = wsb + off;
    off += bytes;
    return p;
  };
  float*   h       = (float*)alloc((size_t)N * 128 * 4);
  bf16_t*  hb      = (bf16_t*)alloc((size_t)N * 128 * 2);
  float*   agg     = (float*)alloc((size_t)N * 128 * 4);   // also 'out' buffer
  float*   logits  = (float*)alloc((size_t)E * 4 * 4);
  float*   pmax    = (float*)alloc(256 * 4 * 4);
  float*   gmax    = (float*)alloc(4 * 4);
  float*   psum    = (float*)alloc(256 * 4 * 4);
  float*   gsuminv = (float*)alloc(4 * 4);
  float*   colpart = (float*)alloc(512 * 128 * 4);
  float*   meanv   = (float*)alloc(128 * 4);
  float*   varv    = (float*)alloc(128 * 4);
  float*   padd    = (float*)alloc((size_t)G * 128 * 4);
  unsigned* pmaxu  = (unsigned*)alloc((size_t)G * 128 * 4);
  float*   pcnt    = (float*)alloc(G * 4);
  const size_t SZ_A1 = 8 * 12 * 512, SZ_M1 = 8 * 12 * 512, SZ_M2 = 8 * 4 * 512,
               SZ_U1 = 8 * 8 * 512, SZ_U2 = 8 * 4 * 512;
  bf16_t* pkA1[4]; bf16_t* pkM1[4]; bf16_t* pkM2[4]; bf16_t* pkU1[4]; bf16_t* pkU2[4];
  for (int l = 0; l < 4; ++l) {
    pkA1[l] = (bf16_t*)alloc(SZ_A1 * 2);
    pkM1[l] = (bf16_t*)alloc(SZ_M1 * 2);
    pkM2[l] = (bf16_t*)alloc(SZ_M2 * 2);
    pkU1[l] = (bf16_t*)alloc(SZ_U1 * 2);
    pkU2[l] = (bf16_t*)alloc(SZ_U2 * 2);
  }

  // Encode nodes + repack weights to bf16 WMMA B-fragments
  node_encode<<<N, 128, 0, stream>>>(x, ne_w, ne_b, h, hb);
  for (int l = 0; l < 4; ++l) {
    pack_wmma_b<<<(int)(SZ_A1 / 256), 256, 0, stream>>>(L(l, 0), pkA1[l], 384, 128);
    pack_wmma_b<<<(int)(SZ_M1 / 256), 256, 0, stream>>>(L(l, 4), pkM1[l], 384, 128);
    pack_wmma_b<<<(int)(SZ_M2 / 256), 256, 0, stream>>>(L(l, 6), pkM2[l], 128, 128);
    pack_wmma_b<<<(int)(SZ_U1 / 256), 256, 0, stream>>>(L(l, 8), pkU1[l], 256, 128);
    pack_wmma_b<<<(int)(SZ_U2 / 256), 256, 0, stream>>>(L(l, 10), pkU2[l], 128, 128);
  }

  for (int l = 0; l < 4; ++l) {
    edge_pass_a<<<E / 16, 256, 0, stream>>>(hb, edge_attr, srcI, dstI, ee_w, ee_b,
                                            pkA1[l], L(l, 1), L(l, 2), L(l, 3), logits);
    attn_partial_max<<<256, 256, 0, stream>>>(logits, pmax, E);
    attn_final_max<<<1, 256, 0, stream>>>(pmax, gmax);
    attn_partial_sum<<<256, 256, 0, stream>>>(logits, gmax, psum, E);
    attn_final_sum<<<1, 256, 0, stream>>>(psum, gsuminv);
    hipMemsetAsync(agg, 0, (size_t)N * 128 * 4, stream);
    edge_pass_b<<<E / 16, 256, 0, stream>>>(hb, edge_attr, srcI, dstI, ee_w, ee_b,
                                            pkM1[l], L(l, 5), pkM2[l], L(l, 7), logits,
                                            gmax, gsuminv, agg);
    node_update<<<N / 16, 256, 0, stream>>>(hb, pkU1[l], L(l, 9), pkU2[l], L(l, 11), agg);
    col_partial_sum<<<512, 128, 0, stream>>>(agg, colpart, N);
    col_final_mean<<<1, 128, 0, stream>>>(colpart, 512, meanv, 1.0f / N);
    col_partial_var<<<512, 128, 0, stream>>>(agg, meanv, L(l, 14), colpart, N);
    col_final_var<<<1, 128, 0, stream>>>(colpart, 512, varv, 1.0f / N);
    norm_residual_ln<<<N / 2, 256, 0, stream>>>(agg, meanv, varv, L(l, 12), L(l, 13),
                                                L(l, 14), L(l, 15), L(l, 16), h, hb);
  }

  hipMemsetAsync(padd, 0, (size_t)G * 128 * 4, stream);
  hipMemsetAsync(pcnt, 0, G * 4, stream);
  fill_u32<<<(G * 128 + 255) / 256, 256, 0, stream>>>(pmaxu, 0x007FFFFFu, G * 128);
  pool_nodes<<<N / 2, 256, 0, stream>>>(h, batch, padd, pmaxu, pcnt);

  HeadW hw[3];
  for (int k = 0; k < 3; ++k) {
    int base = 74 + k * 6;
    hw[k].w1 = (const float*)d_in[base + 0];
    hw[k].b1 = (const float*)d_in[base + 1];
    hw[k].w2 = (const float*)d_in[base + 2];
    hw[k].b2 = (const float*)d_in[base + 3];
    hw[k].w3 = (const float*)d_in[base + 4];
    hw[k].b3 = (const float*)d_in[base + 5];
  }
  heads_kernel<<<G, 128, 0, stream>>>(padd, pmaxu, pcnt, hw[0], hw[1], hw[2],
                                      (float*)d_out);
}
